// ResidualDynamicAMPhaseAttention_62208306316041
// MI455X (gfx1250) — matrix operand order repaired and checked by reading, not compiled, not run on hardware
//
#include <hip/hip_runtime.h>
#include <hip/hip_bf16.h>
#include <math.h>

#define B_   2
#define L_   2048
#define D_   1024
#define H_   16
#define S_   16
#define HD_  64
#define DH_  512            // D/2
#define BL_  (B_*L_)        // 4096
#define BH_  (B_*H_)        // 32
#define QKF_ 96             // hd + 2*S augmented feature width

typedef __attribute__((ext_vector_type(16))) __bf16 v16bf;
typedef __attribute__((ext_vector_type(8)))  __bf16 v8bf;
typedef __attribute__((ext_vector_type(8)))  float  v8f;

__device__ __forceinline__ float softplus_f(float x) {
    return (x > 20.f) ? x : log1pf(expf(x));
}
__device__ __forceinline__ __bf16 tobf(float x) { return (__bf16)x; }

// ---- CDNA5 async global->LDS copy (ASYNCcnt-tracked) ----------------------
__device__ __forceinline__ void async_b128(void* lds, const void* gaddr) {
    uint32_t l = (uint32_t)(uintptr_t)lds;  // low 32 bits = LDS offset
    asm volatile("global_load_async_to_lds_b128 %0, %1, off"
                 :: "v"(l), "v"(gaddr) : "memory");
}
__device__ __forceinline__ void wait_async0() {
    asm volatile("s_wait_asynccnt 0" ::: "memory");
}

// ---- WMMA fragment loaders (ISA 7.12.2 layouts, wave32) -------------------
__device__ __forceinline__ v16bf load_a_frag(const __bf16* lds, int ld, int row0,
                                             int kb, int lane) {
    int l = lane & 15, hi = lane >> 4;
    const __bf16* p = lds + (size_t)(row0 + l) * ld + kb;
    v8bf x0 = *(const v8bf*)(p + hi * 8);
    v8bf x1 = *(const v8bf*)(p + 16 + hi * 8);
    v16bf a;
#pragma unroll
    for (int i = 0; i < 8; i++) { a[i] = x0[i]; a[8 + i] = x1[i]; }
    return a;
}
__device__ __forceinline__ v16bf load_b_frag(const __bf16* lds, int ld, int n0,
                                             int kb, int lane) {
    int n = lane & 15, hi = lane >> 4;
    const __bf16* p = lds + (size_t)(n0 + n) * ld + kb;
    v8bf x0 = *(const v8bf*)(p + hi * 8);
    v8bf x1 = *(const v8bf*)(p + 16 + hi * 8);
    v16bf b;
#pragma unroll
    for (int i = 0; i < 8; i++) { b[i] = x0[i]; b[8 + i] = x1[i]; }
    return b;
}
#define WMMA_BF16(a, b, c) \
    __builtin_amdgcn_wmma_f32_16x16x32_bf16(false, (a), false, (b), (short)0, (c), false, false)

// ======================= packing kernels ====================================
// hidden f32 [M][K] -> bf16 [M][K]
__global__ void k_pack_a(const float* __restrict__ in, __bf16* __restrict__ out, int n) {
    int i = (blockIdx.x * blockDim.x + threadIdx.x) * 8;
    if (i >= n) return;
    const float4* p = (const float4*)(in + i);
    float4 a = p[0], b = p[1];
    v8bf o;
    o[0] = tobf(a.x); o[1] = tobf(a.y); o[2] = tobf(a.z); o[3] = tobf(a.w);
    o[4] = tobf(b.x); o[5] = tobf(b.y); o[6] = tobf(b.z); o[7] = tobf(b.w);
    *(v8bf*)(out + i) = o;
}
// W f32 [K][N] -> bf16 N-major Wt[N][K], LDS-tiled transpose (coalesced both ways)
__global__ __launch_bounds__(256) void k_pack_wt(const float* __restrict__ W,
                                                 __bf16* __restrict__ Wt) {
    __shared__ float T[64][65];
    const int tid = threadIdx.x;
    const int k0 = blockIdx.x * 64, n0 = blockIdx.y * 64;
    for (int i = tid; i < 64 * 64; i += 256) {
        int r = i >> 6, c = i & 63;
        T[r][c] = W[(size_t)(k0 + r) * D_ + n0 + c];
    }
    __syncthreads();
    for (int i = tid; i < 64 * 64; i += 256) {
        int n = i >> 6, k = i & 63;
        Wt[(size_t)(n0 + n) * D_ + k0 + k] = tobf(T[k][n]);
    }
}

// ======================= small prep kernels =================================
__global__ void k_ctx_mean(const float* __restrict__ hs, float* __restrict__ cm) {
    int i = blockIdx.x * blockDim.x + threadIdx.x;
    if (i >= B_ * D_) return;
    int b = i / D_, d = i % D_;
    const float* p = hs + (size_t)b * L_ * D_ + d;
    float s = 0.f;
    for (int l = 0; l < L_; l++) s += p[(size_t)l * D_];
    cm[i] = s * (1.f / L_);
}

__global__ void k_mlp1(const float* __restrict__ cm, const float* __restrict__ w1,
                       const float* __restrict__ b1, float* __restrict__ hdn) {
    int i = blockIdx.x * blockDim.x + threadIdx.x;
    if (i >= B_ * DH_) return;
    int b = i / DH_, j = i % DH_;
    const float* c = cm + b * D_;
    float s = b1[j];
    for (int d = 0; d < D_; d++) s += c[d] * w1[(size_t)d * DH_ + j];
    hdn[i] = 0.5f * s * (1.f + erff(s * 0.70710678118654752f));  // exact gelu
}

__global__ void k_mlp2(const float* __restrict__ hdn, const float* __restrict__ w2,
                       const float* __restrict__ b2, float* __restrict__ deltas) {
    int i = blockIdx.x * blockDim.x + threadIdx.x;
    if (i >= B_ * 512) return;
    int b = i / 512, o = i % 512;
    const float* hh = hdn + b * DH_;
    float s = b2[o];
    for (int j = 0; j < DH_; j++) s += hh[j] * w2[(size_t)j * 512 + o];
    deltas[i] = s;
}

__global__ void k_band(const float* __restrict__ deltas, const float* __restrict__ bgamma,
                       const float* __restrict__ blogit, float* __restrict__ gamma_pos,
                       float* __restrict__ bwsqrt) {
    int i = threadIdx.x;
    if (i >= BH_) return;
    int b = i / H_, h = i % H_;
    float lg[S_];
    float mx = -1e30f;
    for (int s = 0; s < S_; s++) {
        float g = bgamma[h * S_ + s] + deltas[b * 512 + h * S_ + s];
        gamma_pos[i * S_ + s] = softplus_f(g);
        float lo = blogit[h * S_ + s] + deltas[b * 512 + 256 + h * S_ + s];
        lg[s] = lo;
        mx = fmaxf(mx, lo);
    }
    float sum = 0.f;
    for (int s = 0; s < S_; s++) { lg[s] = expf(lg[s] - mx); sum += lg[s]; }
    float inv = 1.f / sum;
    for (int s = 0; s < S_; s++) bwsqrt[i * S_ + s] = sqrtf(lg[s] * inv + 1e-8f);
}

__global__ void k_magprep(const float* __restrict__ mag, const float* __restrict__ gamma_pos,
                          float* __restrict__ mag_pl, float* __restrict__ scale) {
    int i = blockIdx.x * blockDim.x + threadIdx.x;
    if (i >= BL_) return;
    int b = i / L_, l = i % L_;
    float mp[S_];
    float sm = 0.f;
    for (int s = 0; s < S_; s++) { mp[s] = mag[((size_t)b * S_ + s) * L_ + l]; sm += mp[s]; }
    float inv = 1.f / (sm * (1.f / S_) + 1e-8f);
    for (int s = 0; s < S_; s++) { mp[s] *= inv; mag_pl[(size_t)i * S_ + s] = mp[s]; }
    for (int h = 0; h < H_; h++) {
        float mix = 0.f;
        for (int s = 0; s < S_; s++) mix += mp[s] * gamma_pos[(b * H_ + h) * S_ + s];
        scale[((size_t)b * H_ + h) * L_ + l] = 1.f + 0.5f * tanhf(mix);
    }
}

__global__ void k_colmean(const float* __restrict__ mag_pl, float* __restrict__ colmean) {
    int i = threadIdx.x;
    if (i >= B_ * S_) return;
    int b = i / S_, s = i % S_;
    float sum = 0.f;
    for (int l = 0; l < L_; l++) sum += mag_pl[((size_t)b * L_ + l) * S_ + s];
    colmean[i] = sum * (1.f / L_);
}

__global__ void k_burst(const float* __restrict__ mag_pl, const float* __restrict__ colmean,
                        float* __restrict__ burst_mix) {
    int i = blockIdx.x * blockDim.x + threadIdx.x;
    if (i >= BL_) return;
    int b = i / L_, l = i % L_;
    float bm = 0.f;
    for (int s = 0; s < S_; s++) {
        float cm  = colmean[b * S_ + s] + 1e-8f;
        float cur = mag_pl[((size_t)b * L_ + l) * S_ + s] / cm;
        float prv = (l > 0) ? mag_pl[((size_t)b * L_ + l - 1) * S_ + s] / cm : 0.f;
        bm += fmaxf(0.f, cur - prv);
    }
    burst_mix[i] = bm;
}

// fill augmented feature columns 64..95 of Qa/Ka with phase features
__global__ void k_aug(const float* __restrict__ cosp, const float* __restrict__ sinp,
                      const float* __restrict__ bwsqrt, const float* __restrict__ pbias,
                      __bf16* __restrict__ Qa, __bf16* __restrict__ Ka) {
    int i = blockIdx.x * blockDim.x + threadIdx.x;  // over B*H*L
    if (i >= B_ * H_ * L_) return;
    int l = i & (L_ - 1), h = (i >> 11) & 15, b = i >> 15;
    float alpha = sqrtf(softplus_f(pbias[h]) * 2.0f);  // sqrt(hd)/sqrt(S) = 2
    __bf16* q = Qa + (size_t)i * QKF_;
    __bf16* k = Ka + (size_t)i * QKF_;
    for (int s = 0; s < S_; s++) {
        float bw = bwsqrt[(b * H_ + h) * S_ + s] * alpha;
        float c  = cosp[((size_t)b * S_ + s) * L_ + l] * bw;
        float sn = sinp[((size_t)b * S_ + s) * L_ + l] * bw;
        q[64 + s] = tobf(c);  k[64 + s] = tobf(c);
        q[80 + s] = tobf(sn); k[80 + s] = tobf(sn);
    }
}

// ======================= QKV projection (WMMA + async double-buffer) ========
__device__ __forceinline__ void gemm_issue(const __bf16* __restrict__ A,
                                           const __bf16* __restrict__ Wt,
                                           __bf16 (*As)[40], __bf16 (*Bs)[40],
                                           int m0, int n0, int k0, int tid) {
    int r = tid >> 1, seg = (tid & 1) * 16;
    async_b128(&As[r][seg], A  + (size_t)(m0 + r) * D_ + k0 + seg);
    async_b128(&Bs[r][seg], Wt + (size_t)(n0 + r) * D_ + k0 + seg);
}

__global__ __launch_bounds__(256) void k_gemm_qkv(
    const __bf16* __restrict__ A, const __bf16* __restrict__ Wt,
    const float* __restrict__ bias, const float* __restrict__ scale,
    const float* __restrict__ burst_mix, const float* __restrict__ burst_w,
    __bf16* __restrict__ out, int mode) {
    __shared__ __bf16 As[2][128][40];
    __shared__ __bf16 Bs[2][128][40];
    const int tid = threadIdx.x, lane = tid & 31, wave = tid >> 5;
    const int m0 = blockIdx.x * 128, n0 = blockIdx.y * 128;
    const int waveM = wave >> 2, waveN = wave & 3;  // 2x4 wave grid, 64x32 per wave
    v8f acc[4][2] = {};

    gemm_issue(A, Wt, As[0], Bs[0], m0, n0, 0, tid);
    wait_async0();
    __syncthreads();

    for (int k0 = 0, it = 0; k0 < D_; k0 += 32, it++) {
        const int cur = it & 1, nxt = cur ^ 1;
        if (k0 + 32 < D_) gemm_issue(A, Wt, As[nxt], Bs[nxt], m0, n0, k0 + 32, tid);
        v16bf af[4], bfr[2];
#pragma unroll
        for (int fm = 0; fm < 4; fm++)
            af[fm] = load_a_frag(&As[cur][0][0], 40, waveM * 64 + fm * 16, 0, lane);
#pragma unroll
        for (int fn = 0; fn < 2; fn++)
            bfr[fn] = load_b_frag(&Bs[cur][0][0], 40, waveN * 32 + fn * 16, 0, lane);
#pragma unroll
        for (int fm = 0; fm < 4; fm++)
#pragma unroll
            for (int fn = 0; fn < 2; fn++)
                acc[fm][fn] = WMMA_BF16(af[fm], bfr[fn], acc[fm][fn]);
        wait_async0();
        __syncthreads();
    }

    const int hi = lane >> 4, nl = lane & 15;
#pragma unroll
    for (int fm = 0; fm < 4; fm++)
#pragma unroll
        for (int fn = 0; fn < 2; fn++)
#pragma unroll
            for (int r = 0; r < 8; r++) {
                int gm = m0 + waveM * 64 + fm * 16 + hi * 8 + r;
                int gn = n0 + waveN * 32 + fn * 16 + nl;
                float v = acc[fm][fn][r] + bias[gn];
                int b = gm >> 11, l = gm & (L_ - 1);
                int h = gn >> 6,  d = gn & 63;
                int bhl = (b * H_ + h) * L_ + l;
                if (mode < 2) {                       // Q/K, 96-wide augmented rows
                    v *= scale[bhl];
                    out[(size_t)bhl * QKF_ + d] = tobf(v);
                } else {                              // V, transposed [bh][d][l]
                    float bw = softplus_f(burst_w[h]);
                    v *= (1.f + bw * burst_mix[b * L_ + l]);
                    out[((size_t)(b * H_ + h) * HD_ + d) * L_ + l] = tobf(v);
                }
            }
}

// ======================= flash attention (WMMA + async double-buffer) =======
__device__ __forceinline__ void attn_issue(const __bf16* __restrict__ Kg,
                                           const __bf16* __restrict__ Vtg,
                                           __bf16 (*Ks)[104], __bf16 (*Vs)[72],
                                           int ct, int tid) {
#pragma unroll
    for (int j = 0; j < 3; j++) {       // K tile: 64 rows x 96 = 384 b128 chunks
        int c = tid + j * 128;
        int r = c / 6, sg = (c % 6) * 16;
        async_b128(&Ks[r][sg], Kg + (size_t)(ct * 64 + r) * QKF_ + sg);
    }
#pragma unroll
    for (int j = 0; j < 2; j++) {       // V tile: 64 d-rows x 64 keys = 256 chunks
        int c = tid + j * 128;
        int d = c >> 2, sg = (c & 3) * 16;
        async_b128(&Vs[d][sg], Vtg + (size_t)d * L_ + ct * 64 + sg);
    }
}

__global__ __launch_bounds__(128) void k_attn(
    const __bf16* __restrict__ Qa, const __bf16* __restrict__ Ka,
    const __bf16* __restrict__ Vtg_all, const float* __restrict__ mask,
    __bf16* __restrict__ ctxo) {
    __shared__ union {
        __bf16 Q[64][104];              // staging for Q (used once)
        __bf16 P[4][16][64];            // per-wave probability staging
    } u;
    __shared__ __bf16 Ks[2][64][104];
    __shared__ __bf16 Vs[2][64][72];    // Vs[d][key]
    const int tid = threadIdx.x, lane = tid & 31, wave = tid >> 5;
    const int rt = blockIdx.x, bh = blockIdx.y;
    const int b = bh >> 4, h = bh & 15;
    const int hi = lane >> 4, nl = lane & 15;

    const __bf16* Qg  = Qa + ((size_t)bh * L_ + rt * 64) * QKF_;
    const __bf16* Kg  = Ka + (size_t)bh * L_ * QKF_;
    const __bf16* Vtg = Vtg_all + (size_t)bh * HD_ * L_;

#pragma unroll
    for (int j = 0; j < 3; j++) {       // async-stage Q tile
        int c = tid + j * 128;
        int r = c / 6, sg = (c % 6) * 16;
        async_b128(&u.Q[r][sg], Qg + (size_t)r * QKF_ + sg);
    }
    attn_issue(Kg, Vtg, Ks[0], Vs[0], 0, tid);
    wait_async0();
    __syncthreads();

    v16bf qf[3];
#pragma unroll
    for (int ks = 0; ks < 3; ks++)
        qf[ks] = load_a_frag(&u.Q[0][0], 104, wave * 16, ks * 32, lane);
    __syncthreads();                    // guard: Q reads done before P overwrites

    float m_[8], l_[8];
    v8f o_[4] = {};
#pragma unroll
    for (int r = 0; r < 8; r++) { m_[r] = -1e30f; l_[r] = 0.f; }

    for (int ct = 0; ct < L_ / 64; ct++) {
        const int cur = ct & 1, nxt = cur ^ 1;
        if (ct + 1 < L_ / 64) attn_issue(Kg, Vtg, Ks[nxt], Vs[nxt], ct + 1, tid);

        v8f s[4] = {};
#pragma unroll
        for (int j = 0; j < 4; j++)
#pragma unroll
            for (int ks = 0; ks < 3; ks++) {
                v16bf kb_ = load_b_frag(&Ks[cur][0][0], 104, j * 16, ks * 32, lane);
                s[j] = WMMA_BF16(qf[ks], kb_, s[j]);
            }
        const float invsq = 0.125f;  // 1/sqrt(64)
#pragma unroll
        for (int j = 0; j < 4; j++) {
            float mk = mask[b * L_ + ct * 64 + j * 16 + nl];
#pragma unroll
            for (int r = 0; r < 8; r++) s[j][r] = s[j][r] * invsq + mk;
        }
        float corr[8], mnew[8];
#pragma unroll
        for (int r = 0; r < 8; r++) {
            float mx = s[0][r];
#pragma unroll
            for (int j = 1; j < 4; j++) mx = fmaxf(mx, s[j][r]);
            mx = fmaxf(mx, __shfl_xor(mx, 1, 16));
            mx = fmaxf(mx, __shfl_xor(mx, 2, 16));
            mx = fmaxf(mx, __shfl_xor(mx, 4, 16));
            mx = fmaxf(mx, __shfl_xor(mx, 8, 16));
            mnew[r] = fmaxf(m_[r], mx);
            corr[r] = expf(m_[r] - mnew[r]);
        }
#pragma unroll
        for (int r = 0; r < 8; r++) {
            float rs = 0.f;
#pragma unroll
            for (int j = 0; j < 4; j++) {
                float p = expf(s[j][r] - mnew[r]);
                u.P[wave][hi * 8 + r][j * 16 + nl] = tobf(p);
                rs += p;
            }
            rs += __shfl_xor(rs, 1, 16);
            rs += __shfl_xor(rs, 2, 16);
            rs += __shfl_xor(rs, 4, 16);
            rs += __shfl_xor(rs, 8, 16);
            l_[r] = l_[r] * corr[r] + rs;
            m_[r] = mnew[r];
        }
#pragma unroll
        for (int oj = 0; oj < 4; oj++)
#pragma unroll
            for (int r = 0; r < 8; r++) o_[oj][r] *= corr[r];
#pragma unroll
        for (int ks2 = 0; ks2 < 2; ks2++) {
            v16bf ap = load_a_frag(&u.P[wave][0][0], 64, 0, ks2 * 32, lane);
#pragma unroll
            for (int oj = 0; oj < 4; oj++) {
                v16bf bv = load_b_frag(&Vs[cur][0][0], 72, oj * 16, ks2 * 32, lane);
                o_[oj] = WMMA_BF16(ap, bv, o_[oj]);
            }
        }
        wait_async0();
        __syncthreads();
    }
#pragma unroll
    for (int r = 0; r < 8; r++) {
        float invl = 1.f / l_[r];
        int gl = rt * 64 + wave * 16 + hi * 8 + r;
#pragma unroll
        for (int oj = 0; oj < 4; oj++) {
            int d = oj * 16 + nl;
            ctxo[((size_t)(b * L_ + gl)) * D_ + h * HD_ + d] = tobf(o_[oj][r] * invl);
        }
    }
}

// ======================= output projection (WMMA + async) ===================
__global__ __launch_bounds__(256) void k_gemm_o(
    const __bf16* __restrict__ Actx, const __bf16* __restrict__ Wt,
    const float* __restrict__ bias, const float* __restrict__ resid,
    float* __restrict__ Y) {
    __shared__ __bf16 As[2][128][40];
    __shared__ __bf16 Bs[2][128][40];
    const int tid = threadIdx.x, lane = tid & 31, wave = tid >> 5;
    const int m0 = blockIdx.x * 128, n0 = blockIdx.y * 128;
    const int waveM = wave >> 2, waveN = wave & 3;
    v8f acc[4][2] = {};

    gemm_issue(Actx, Wt, As[0], Bs[0], m0, n0, 0, tid);
    wait_async0();
    __syncthreads();

    for (int k0 = 0, it = 0; k0 < D_; k0 += 32, it++) {
        const int cur = it & 1, nxt = cur ^ 1;
        if (k0 + 32 < D_) gemm_issue(Actx, Wt, As[nxt], Bs[nxt], m0, n0, k0 + 32, tid);
        v16bf af[4], bfr[2];
#pragma unroll
        for (int fm = 0; fm < 4; fm++)
            af[fm] = load_a_frag(&As[cur][0][0], 40, waveM * 64 + fm * 16, 0, lane);
#pragma unroll
        for (int fn = 0; fn < 2; fn++)
            bfr[fn] = load_b_frag(&Bs[cur][0][0], 40, waveN * 32 + fn * 16, 0, lane);
#pragma unroll
        for (int fm = 0; fm < 4; fm++)
#pragma unroll
            for (int fn = 0; fn < 2; fn++)
                acc[fm][fn] = WMMA_BF16(af[fm], bfr[fn], acc[fm][fn]);
        wait_async0();
        __syncthreads();
    }
    const int hi = lane >> 4, nl = lane & 15;
#pragma unroll
    for (int fm = 0; fm < 4; fm++)
#pragma unroll
        for (int fn = 0; fn < 2; fn++)
#pragma unroll
            for (int r = 0; r < 8; r++) {
                int gm = m0 + waveM * 64 + fm * 16 + hi * 8 + r;
                int gn = n0 + waveN * 32 + fn * 16 + nl;
                Y[(size_t)gm * D_ + gn] =
                    acc[fm][fn][r] + bias[gn] + resid[(size_t)gm * D_ + gn];
            }
}

// ======================= layernorm (in-place) ===============================
__global__ __launch_bounds__(256) void k_ln(float* __restrict__ Y,
                                            const float* __restrict__ g,
                                            const float* __restrict__ bb) {
    const int row = blockIdx.x, tid = threadIdx.x;
    float* p = Y + (size_t)row * D_;
    float s = 0.f, s2 = 0.f;
    for (int c = tid; c < D_; c += 256) { float x = p[c]; s += x; s2 += x * x; }
    for (int off = 1; off < 32; off <<= 1) {
        s  += __shfl_xor(s, off, 32);
        s2 += __shfl_xor(s2, off, 32);
    }
    __shared__ float sh[16];
    int wave = tid >> 5, lane = tid & 31;
    if (lane == 0) { sh[wave] = s; sh[8 + wave] = s2; }
    __syncthreads();
    if (tid == 0) {
        float a = 0.f, b2 = 0.f;
        for (int w = 0; w < 8; w++) { a += sh[w]; b2 += sh[8 + w]; }
        sh[0] = a; sh[8] = b2;
    }
    __syncthreads();
    float mu  = sh[0] * (1.f / D_);
    float var = sh[8] * (1.f / D_) - mu * mu;
    float inv = rsqrtf(var + 1e-12f);
    for (int c = tid; c < D_; c += 256)
        p[c] = (p[c] - mu) * inv * g[c] + bb[c];
}

// ======================= launcher ==========================================
static inline size_t alignup256(size_t x) { return (x + 255) & ~(size_t)255; }

extern "C" void kernel_launch(void* const* d_in, const int* in_sizes, int n_in,
                              void* d_out, int out_size, void* d_ws, size_t ws_size,
                              hipStream_t stream) {
    const float* hs    = (const float*)d_in[0];
    const float* maskp = (const float*)d_in[1];
    const float* cosp  = (const float*)d_in[2];
    const float* sinp  = (const float*)d_in[3];
    const float* mag   = (const float*)d_in[4];
    const float* wq = (const float*)d_in[5],  *bq = (const float*)d_in[6];
    const float* wk = (const float*)d_in[7],  *bk = (const float*)d_in[8];
    const float* wv = (const float*)d_in[9],  *bv = (const float*)d_in[10];
    const float* wo = (const float*)d_in[11], *bo = (const float*)d_in[12];
    const float* ln_g = (const float*)d_in[13], *ln_b = (const float*)d_in[14];
    const float* bbl  = (const float*)d_in[15];
    const float* pbias = (const float*)d_in[16];
    const float* bgam  = (const float*)d_in[17];
    const float* w1 = (const float*)d_in[18], *b1 = (const float*)d_in[19];
    const float* w2 = (const float*)d_in[20], *b2 = (const float*)d_in[21];
    const float* bw = (const float*)d_in[22];

    char* ws = (char*)d_ws;
    size_t off = 0;
    auto take = [&](size_t bytes) -> void* {
        void* p = ws + off;
        off = alignup256(off + bytes);
        return p;
    };
    float* ctxm      = (float*)take((size_t)B_ * D_ * 4);
    float* hdn       = (float*)take((size_t)B_ * DH_ * 4);
    float* deltas    = (float*)take((size_t)B_ * 512 * 4);
    float* gamma_pos = (float*)take((size_t)BH_ * S_ * 4);
    float* bwsq      = (float*)take((size_t)BH_ * S_ * 4);
    float* mag_pl    = (float*)take((size_t)BL_ * S_ * 4);
    float* colmean   = (float*)take((size_t)B_ * S_ * 4);
    float* burstmx   = (float*)take((size_t)BL_ * 4);
    float* scale     = (float*)take((size_t)BH_ * L_ * 4);
    __bf16* Abf  = (__bf16*)take((size_t)BL_ * D_ * 2);       // packed hidden
    __bf16* Wqt  = (__bf16*)take((size_t)D_ * D_ * 2);
    __bf16* Wkt  = (__bf16*)take((size_t)D_ * D_ * 2);
    __bf16* Wvt  = (__bf16*)take((size_t)D_ * D_ * 2);
    __bf16* Wot  = (__bf16*)take((size_t)D_ * D_ * 2);
    __bf16* Qa   = (__bf16*)take((size_t)BH_ * L_ * QKF_ * 2);
    __bf16* Ka   = (__bf16*)take((size_t)BH_ * L_ * QKF_ * 2);
    __bf16* Vtg  = (__bf16*)take((size_t)BH_ * HD_ * L_ * 2); // V transposed [bh][d][l]
    __bf16* ctxo = (__bf16*)take((size_t)BL_ * D_ * 2);
    (void)ws_size; (void)n_in; (void)in_sizes; (void)out_size;

    // one-time packing to bf16 (WMMA-friendly layouts)
    k_pack_a<<<(BL_ * D_ / 8 + 255) / 256, 256, 0, stream>>>(hs, Abf, BL_ * D_);
    dim3 gt(D_ / 64, D_ / 64);
    k_pack_wt<<<gt, 256, 0, stream>>>(wq, Wqt);
    k_pack_wt<<<gt, 256, 0, stream>>>(wk, Wkt);
    k_pack_wt<<<gt, 256, 0, stream>>>(wv, Wvt);
    k_pack_wt<<<gt, 256, 0, stream>>>(wo, Wot);

    k_ctx_mean<<<(B_ * D_ + 255) / 256, 256, 0, stream>>>(hs, ctxm);
    k_mlp1<<<(B_ * DH_ + 255) / 256, 256, 0, stream>>>(ctxm, w1, b1, hdn);
    k_mlp2<<<(B_ * 512 + 255) / 256, 256, 0, stream>>>(hdn, w2, b2, deltas);
    k_band<<<1, 32, 0, stream>>>(deltas, bgam, bbl, gamma_pos, bwsq);
    k_magprep<<<(BL_ + 255) / 256, 256, 0, stream>>>(mag, gamma_pos, mag_pl, scale);
    k_colmean<<<1, 32, 0, stream>>>(mag_pl, colmean);
    k_burst<<<(BL_ + 255) / 256, 256, 0, stream>>>(mag_pl, colmean, burstmx);
    k_aug<<<(B_ * H_ * L_ + 255) / 256, 256, 0, stream>>>(cosp, sinp, bwsq, pbias, Qa, Ka);

    dim3 gq(BL_ / 128, D_ / 128);
    k_gemm_qkv<<<gq, 256, 0, stream>>>(Abf, Wqt, bq, scale, burstmx, bw, Qa, 0);
    k_gemm_qkv<<<gq, 256, 0, stream>>>(Abf, Wkt, bk, scale, burstmx, bw, Ka, 1);
    k_gemm_qkv<<<gq, 256, 0, stream>>>(Abf, Wvt, bv, scale, burstmx, bw, Vtg, 2);

    k_attn<<<dim3(L_ / 64, BH_), 128, 0, stream>>>(Qa, Ka, Vtg, maskp, ctxo);

    k_gemm_o<<<gq, 256, 0, stream>>>(ctxo, Wot, bo, hs, (float*)d_out);
    k_ln<<<BL_, 256, 0, stream>>>((float*)d_out, ln_g, ln_b);
}
